// SAGE_32822140076407
// MI455X (gfx1250) — matrix operand order repaired and checked by reading, not compiled, not run on hardware
//
#include <hip/hip_runtime.h>

// GraphSAGE 3-layer forward for MI455X (gfx1250, wave32, WMMA).
// Aggregation: f16 gather + GLOBAL_ATOMIC_PK_ADD_F16 scatter (~0.5GB/layer).
// Dense linear layers: v_wmma_f32_16x16x32_f16 (f16 in, f32 accumulate), with
// mean-normalization fused into the A-fragment load (precomputed f16 reciprocal
// counts) and ReLU/dropout fused into the epilogue, which emits the next
// layer's f16 feature buffer directly.

#define GN 100000
#define GE 800000
#define DROP_P 0.3f

typedef __attribute__((ext_vector_type(16))) _Float16 v16h;
typedef __attribute__((ext_vector_type(8)))  _Float16 v8h;
typedef __attribute__((ext_vector_type(8)))  float    v8f;

// ---------------- f32 -> f16 conversion (weights, initial features) ----------
__global__ void f32_to_f16_kernel(const float* __restrict__ in,
                                  _Float16* __restrict__ out, int n) {
  int i = blockIdx.x * blockDim.x + threadIdx.x;
  if (i < n) out[i] = (_Float16)in[i];
}

// ---------------- per-node reciprocal degree (computed once) -----------------
__global__ void inv_cnt_kernel(const float* __restrict__ cnt,
                               _Float16* __restrict__ inv16, int n) {
  int i = blockIdx.x * blockDim.x + threadIdx.x;
  if (i < n) inv16[i] = (_Float16)(1.0f / fmaxf(cnt[i], 1.0f));
}

// ---------------- edge scatter-add: f16 gather + packed-f16 atomics ----------
// One wave32 per edge; each lane moves 4 channels (4*32 = 128) with one 8B load
// and two global_atomic_pk_add_f16.
__device__ __forceinline__ void pk_atomic_add_f16(_Float16* p, unsigned packed) {
  unsigned long long addr = (unsigned long long)p;
  asm volatile("global_atomic_pk_add_f16 %0, %1, off scope:SCOPE_DEV"
               :: "v"(addr), "v"(packed) : "memory");
}

__global__ void scatter_kernel(const _Float16* __restrict__ x16,
                               const int* __restrict__ src,
                               const int* __restrict__ dst,
                               _Float16* __restrict__ agg16,
                               float* __restrict__ cnt,
                               int nedges, int addCnt) {
  int wave = (blockIdx.x * blockDim.x + threadIdx.x) >> 5;
  int lane = threadIdx.x & 31;
  if (wave >= nedges) return;            // wave-uniform
  int s = src[wave];
  int d = dst[wave];
  uint2 v = *(const uint2*)(x16 + (size_t)s * 128 + lane * 4);  // 4 halves
  _Float16* a = agg16 + (size_t)d * 128 + lane * 4;
  pk_atomic_add_f16(a, v.x);
  pk_atomic_add_f16(a + 2, v.y);
  if (addCnt && lane == 0) atomicAdd(cnt + d, 1.0f);
}

// ---------------- deterministic dropout hash ----------------
__device__ __forceinline__ unsigned hash_u32(unsigned x) {
  x ^= x >> 16; x *= 0x7feb352dU;
  x ^= x >> 15; x *= 0x846ca68bU;
  x ^= x >> 16;
  return x;
}

// ---------------- WMMA GEMM ----------------
// out = (agg*inv)@Wl + bias + x@Wr  (+relu+dropout), K = 128.
// One wave per 16x16 output tile (looped over row tiles); grid.y = COUT/16.
// B fragments (both weights, all 4 K-chunks) held in registers.
template <int COUT, bool RELU_DROP, bool HALF_OUT>
__global__ void sage_gemm_kernel(const _Float16* __restrict__ agg16,
                                 const _Float16* __restrict__ inv16,
                                 const _Float16* __restrict__ xh,
                                 const _Float16* __restrict__ wl,
                                 const _Float16* __restrict__ wr,
                                 const float* __restrict__ bias,
                                 void* __restrict__ outv,
                                 int rowTiles, unsigned seed) {
  const int lane = threadIdx.x & 31;
  const int wavesPerBlock = blockDim.x >> 5;
  const int waveId = blockIdx.x * wavesPerBlock + (threadIdx.x >> 5);
  const int nWaves = gridDim.x * wavesPerBlock;
  const int col0 = blockIdx.y * 16;
  const int n  = lane & 15;   // B/C/D column within tile; also A row M = lane%16
  const int hf = lane >> 4;   // half-wave select
  float* outf = (float*)outv;
  _Float16* outh = (_Float16*)outv;

  // B fragment layout (32x16 per K-chunk): lanes 0-15 hold K=kk*32+{0..7,16..23},
  // lanes 16-31 hold K=kk*32+{8..15,24..31}, column = lane%16; b[j] = W[kbase+j+(j>=8)*8][col].
  v16h blf[4], brf[4];
#pragma unroll
  for (int kk = 0; kk < 4; ++kk) {
    const int kb = kk * 32 + hf * 8;
#pragma unroll
    for (int j = 0; j < 16; ++j) {
      blf[kk][j] = wl[(kb + j + (j >= 8) * 8) * COUT + col0 + n];
      brf[kk][j] = wr[(kb + j + (j >= 8) * 8) * COUT + col0 + n];
    }
  }
  const float bv = bias[col0 + n];

  for (int r = waveId; r < rowTiles; r += nWaves) {
    const int arow = r * 16 + n;                      // A row for this lane
    const size_t abase = (size_t)arow * 128;
    const _Float16 hinv = inv16[arow];                // precomputed 1/max(deg,1)
    v8f acc = {};
#pragma unroll
    for (int kk = 0; kk < 4; ++kk) {
      // A 16-bit layout: half hf -> halves [kofs..kofs+7] and [kofs+16..kofs+23]
      const int kofs = kk * 32 + hf * 8;
      v8h lo = *(const v8h*)(agg16 + abase + kofs);
      v8h hi = *(const v8h*)(agg16 + abase + kofs + 16);
      v16h a;
#pragma unroll
      for (int j = 0; j < 8; ++j) { a[j] = lo[j] * hinv; a[8 + j] = hi[j] * hinv; }
      acc = __builtin_amdgcn_wmma_f32_16x16x32_f16(false, a, false, blf[kk],
                                                   (short)0, acc, false, false);
      v8h lo2 = *(const v8h*)(xh + abase + kofs);
      v8h hi2 = *(const v8h*)(xh + abase + kofs + 16);
      v16h ax;
#pragma unroll
      for (int j = 0; j < 8; ++j) { ax[j] = lo2[j]; ax[8 + j] = hi2[j]; }
      acc = __builtin_amdgcn_wmma_f32_16x16x32_f16(false, ax, false, brf[kk],
                                                   (short)0, acc, false, false);
    }
    // Epilogue: lane covers column col0+n, rows r*16 + hf*8 + d (d = VGPR index)
#pragma unroll
    for (int d = 0; d < 8; ++d) {
      const int grow = r * 16 + hf * 8 + d;
      float v = acc[d] + bv;
      if (RELU_DROP) {
        v = fmaxf(v, 0.0f);
        unsigned idx = (unsigned)grow * (unsigned)COUT + (unsigned)(col0 + n);
        float u = (float)hash_u32(idx ^ seed) * 2.3283064365386963e-10f;
        v = (u >= DROP_P) ? v * (1.0f / (1.0f - DROP_P)) : 0.0f;
      }
      const size_t oidx = (size_t)grow * COUT + col0 + n;
      if (HALF_OUT) outh[oidx] = (_Float16)v;
      else          outf[oidx] = v;
    }
  }
}

extern "C" void kernel_launch(void* const* d_in, const int* in_sizes, int n_in,
                              void* d_out, int out_size, void* d_ws, size_t ws_size,
                              hipStream_t stream) {
  (void)in_sizes; (void)n_in; (void)out_size; (void)ws_size;
  const float* x   = (const float*)d_in[0];
  const int*   ei  = (const int*)d_in[1];       // [2,E]: row 0 = src, row 1 = dst
  const int*   src = ei;
  const int*   dst = ei + GE;
  const float* Wl0 = (const float*)d_in[2];
  const float* bl0 = (const float*)d_in[3];
  const float* Wr0 = (const float*)d_in[4];
  const float* Wl1 = (const float*)d_in[5];
  const float* bl1 = (const float*)d_in[6];
  const float* Wr1 = (const float*)d_in[7];
  const float* Wl2 = (const float*)d_in[8];
  const float* bl2 = (const float*)d_in[9];
  const float* Wr2 = (const float*)d_in[10];
  float* outp = (float*)d_out;

  // ---- workspace carve-up (256B aligned) ----
  char* ws = (char*)d_ws;
  size_t off = 0;
  auto carve = [&](size_t bytes) -> void* {
    void* p = ws + off;
    off += (bytes + 255) & ~(size_t)255;
    return p;
  };
  _Float16* agg16 = (_Float16*)carve((size_t)GN * 128 * sizeof(_Float16));
  float*    cnt   = (float*)carve((size_t)GN * sizeof(float));
  _Float16* inv16 = (_Float16*)carve((size_t)GN * sizeof(_Float16));
  _Float16* x16a  = (_Float16*)carve((size_t)GN * 128 * sizeof(_Float16));
  _Float16* x16b  = (_Float16*)carve((size_t)GN * 128 * sizeof(_Float16));
  _Float16* wl0h  = (_Float16*)carve(128 * 128 * sizeof(_Float16));
  _Float16* wr0h  = (_Float16*)carve(128 * 128 * sizeof(_Float16));
  _Float16* wl1h  = (_Float16*)carve(128 * 128 * sizeof(_Float16));
  _Float16* wr1h  = (_Float16*)carve(128 * 128 * sizeof(_Float16));
  _Float16* wl2h  = (_Float16*)carve(128 * 64 * sizeof(_Float16));
  _Float16* wr2h  = (_Float16*)carve(128 * 64 * sizeof(_Float16));

  // ---- convert weights + initial features to f16 ----
  f32_to_f16_kernel<<<(16384 + 255) / 256, 256, 0, stream>>>(Wl0, wl0h, 16384);
  f32_to_f16_kernel<<<(16384 + 255) / 256, 256, 0, stream>>>(Wr0, wr0h, 16384);
  f32_to_f16_kernel<<<(16384 + 255) / 256, 256, 0, stream>>>(Wl1, wl1h, 16384);
  f32_to_f16_kernel<<<(16384 + 255) / 256, 256, 0, stream>>>(Wr1, wr1h, 16384);
  f32_to_f16_kernel<<<(8192 + 255) / 256, 256, 0, stream>>>(Wl2, wl2h, 8192);
  f32_to_f16_kernel<<<(8192 + 255) / 256, 256, 0, stream>>>(Wr2, wr2h, 8192);
  f32_to_f16_kernel<<<(GN * 128) / 256, 256, 0, stream>>>(x, x16a, GN * 128);

  const int scatterBlocks = GE / 8;   // 8 waves (edges) per 256-thread block
  const int rowTiles      = GN / 16;  // 6250 exactly
  const size_t aggBytes   = (size_t)GN * 128 * sizeof(_Float16);

  // ================= layer 0: x16a -> x16b (128->128, relu+dropout) ==========
  hipMemsetAsync(agg16, 0, aggBytes, stream);
  hipMemsetAsync(cnt, 0, (size_t)GN * sizeof(float), stream);
  scatter_kernel<<<scatterBlocks, 256, 0, stream>>>(x16a, src, dst, agg16, cnt, GE, 1);
  inv_cnt_kernel<<<(GN + 255) / 256, 256, 0, stream>>>(cnt, inv16, GN);
  sage_gemm_kernel<128, true, true><<<dim3(128, 8), 256, 0, stream>>>(
      agg16, inv16, x16a, wl0h, wr0h, bl0, x16b, rowTiles, 0x243F6A88u);

  // ================= layer 1: x16b -> x16a (128->128, relu+dropout) ==========
  hipMemsetAsync(agg16, 0, aggBytes, stream);
  scatter_kernel<<<scatterBlocks, 256, 0, stream>>>(x16b, src, dst, agg16, cnt, GE, 0);
  sage_gemm_kernel<128, true, true><<<dim3(128, 8), 256, 0, stream>>>(
      agg16, inv16, x16b, wl1h, wr1h, bl1, x16a, rowTiles, 0x85A308D3u);

  // ================= layer 2: x16a -> d_out (128->64, f32, plain) ============
  hipMemsetAsync(agg16, 0, aggBytes, stream);
  scatter_kernel<<<scatterBlocks, 256, 0, stream>>>(x16a, src, dst, agg16, cnt, GE, 0);
  sage_gemm_kernel<64, false, false><<<dim3(128, 4), 256, 0, stream>>>(
      agg16, inv16, x16a, wl2h, wr2h, bl2, outp, rowTiles, 0u);
}
